// MultiHeadAttention_72009421685206
// MI455X (gfx1250) — compile-verified, matrix operand order checked by reading, not run on hardware
//
#include <hip/hip_runtime.h>
#include <hip/hip_bf16.h>

// ---------------------------------------------------------------------------
// Multi-head attention forward, CDNA5 (gfx1250), bf16 WMMA pipeline with
// async global->LDS staging (ASYNCcnt) for the shared B-operand slabs.
//   out  = (softmax(QWq (KWk)^T / sqrt(dk)) (VWv)) Wo^T  (+ biases)
// d_out = [ out : 2*2048*1024 f32 | attn : 2*16*2048*2048 f32 ]
// ---------------------------------------------------------------------------

typedef __attribute__((ext_vector_type(16))) __bf16 v16bf;
typedef __attribute__((ext_vector_type(8)))  __bf16 v8bf;
typedef __attribute__((ext_vector_type(8)))  float  v8f;

#define D_MODEL 1024
#define N_HEAD  16
#define D_K     64
#define SEQ     2048
#define BATCH   2
#define ROWS    (BATCH * SEQ)        // 4096

// ---------------------------------------------------------------------------
// Fragment loaders.  CDNA5 16-bit A/B 16x32 layout (ISA 7.12.2):
//   lanes 0-15  : row M = lane,    K = {0..7} then {16..23}
//   lanes 16-31 : row M = lane-16, K = {8..15} then {24..31}
// ---------------------------------------------------------------------------
__device__ __forceinline__ v16bf load_frag_bf16(const __bf16* __restrict__ p, int ld) {
    const int lane = threadIdx.x & 31;
    const int row  = lane & 15;
    const int kofs = (lane >> 4) << 3;          // 0 or 8
    const __bf16* q = p + row * ld + kofs;
    v8bf lo = *(const v8bf*)(q);
    v8bf hi = *(const v8bf*)(q + 16);
    v16bf f;
#pragma unroll
    for (int i = 0; i < 8; ++i) { f[i] = lo[i]; f[i + 8] = hi[i]; }
    return f;
}

// Same layout from an LDS-resident 64x32 slab (row stride 32 elements).
__device__ __forceinline__ v16bf load_frag_smem(const __bf16* p) {
    const int lane = threadIdx.x & 31;
    const int row  = lane & 15;
    const int kofs = (lane >> 4) << 3;
    const __bf16* q = p + row * 32 + kofs;
    v8bf lo = *(const v8bf*)(q);
    v8bf hi = *(const v8bf*)(q + 16);
    v16bf f;
#pragma unroll
    for (int i = 0; i < 8; ++i) { f[i] = lo[i]; f[i + 8] = hi[i]; }
    return f;
}

// f32 source (attention probabilities), converted to bf16 in-regs.
__device__ __forceinline__ v16bf load_frag_f32(const float* __restrict__ p, int ld) {
    const int lane = threadIdx.x & 31;
    const int row  = lane & 15;
    const int kofs = (lane >> 4) << 3;
    const float* q = p + row * ld + kofs;
    v8f lo = *(const v8f*)(q);
    v8f hi = *(const v8f*)(q + 16);
    v16bf f;
#pragma unroll
    for (int i = 0; i < 8; ++i) { f[i] = (__bf16)lo[i]; f[i + 8] = (__bf16)hi[i]; }
    return f;
}

__device__ __forceinline__ v8f wmma_bf16(v16bf a, v16bf b, v8f c) {
    return __builtin_amdgcn_wmma_f32_16x16x32_bf16(
        false, a, false, b, (short)0, c, false, false);
}

// ---------------------------------------------------------------------------
// CDNA5 async global->LDS copy (16 bytes per lane, tracked by ASYNCcnt).
// dst_smem is an LDS-backed pointer: the generic address' low 32 bits are the
// LDS byte offset (shared-aperture base has zero low bits).
// ---------------------------------------------------------------------------
__device__ __forceinline__ void async_copy_b128(const __bf16* __restrict__ src,
                                                __bf16* dst_smem) {
    unsigned           dst = (unsigned)(unsigned long long)(uintptr_t)dst_smem;
    unsigned long long s   = (unsigned long long)(uintptr_t)src;
    asm volatile("global_load_async_to_lds_b128 %0, %1, off"
                 :: "v"(dst), "v"(s) : "memory");
}

__device__ __forceinline__ void wait_async0() {
    asm volatile("s_wait_asynccnt 0x0" ::: "memory");
}

// Stage one 64x32 bf16 slab (4 KB): 256 threads x one async b128 each.
// Source rows are K-contiguous with stride ldb.
__device__ __forceinline__ void stage_slab(const __bf16* __restrict__ src,
                                           int ldb, int k, __bf16* smem) {
    const int t     = threadIdx.x;
    const int row   = t >> 2;            // 0..63
    const int chunk = (t & 3) << 3;      // 0,8,16,24 elements
    async_copy_b128(src + (size_t)row * ldb + k + chunk,
                    smem + row * 32 + chunk);
}

// ---------------------------------------------------------------------------
// f32 -> bf16 conversion (grid-stride)
// ---------------------------------------------------------------------------
__global__ void __launch_bounds__(256)
convert_kernel(const float* __restrict__ src, __bf16* __restrict__ dst, int n) {
    int i      = blockIdx.x * blockDim.x + threadIdx.x;
    int stride = gridDim.x * blockDim.x;
    for (; i < n; i += stride) dst[i] = (__bf16)src[i];
}

// ---------------------------------------------------------------------------
// Projection GEMM: Y[m,n] = sum_k X[m,k] * W[n,k] + bias[n]
// M=4096, N=1024, K=1024.  Block = 8 waves covering 512 rows x SAME 64 cols;
// the shared W slab is double-buffered in LDS via async copies.
// MODE 0: store to head layout    [b,h,s,d]  (Q, K)
// MODE 1: store to head-transpose [b,h,d,s]  (V -> makes ctx GEMM K-contig)
// ---------------------------------------------------------------------------
template <int MODE>
__global__ void __launch_bounds__(256)
proj_kernel(const __bf16* __restrict__ X, const __bf16* __restrict__ W,
            const float* __restrict__ bias, __bf16* __restrict__ out) {
    __shared__ __bf16 smemB[2][64 * 32];

    const int wave   = threadIdx.x >> 5;
    const int blockM = blockIdx.x >> 4;           // 8 M-blocks of 512 rows
    const int blockN = blockIdx.x & 15;           // 16 N-blocks of 64 cols
    const int M0 = blockM * 512 + wave * 64;
    const int N0 = blockN * 64;

    const __bf16* Wb = W + (size_t)N0 * D_MODEL;
    const int NK = D_MODEL / 32;                  // 32 k-steps

    stage_slab(Wb, D_MODEL, 0, smemB[0]);
    wait_async0();
    __syncthreads();

    v8f acc[4][4] = {};
    for (int kk = 0; kk < NK; ++kk) {
        const int cur = kk & 1;
        if (kk + 1 < NK) stage_slab(Wb, D_MODEL, (kk + 1) * 32, smemB[1 - cur]);

        const int k = kk * 32;
        v16bf a[4], b[4];
#pragma unroll
        for (int i = 0; i < 4; ++i)
            a[i] = load_frag_bf16(X + (size_t)(M0 + i * 16) * D_MODEL + k, D_MODEL);
#pragma unroll
        for (int j = 0; j < 4; ++j)
            b[j] = load_frag_smem(&smemB[cur][j * 16 * 32]);
#pragma unroll
        for (int i = 0; i < 4; ++i)
#pragma unroll
            for (int j = 0; j < 4; ++j)
                acc[i][j] = wmma_bf16(a[i], b[j], acc[i][j]);

        if (kk + 1 < NK) { wait_async0(); __syncthreads(); }
    }

    const int lane = threadIdx.x & 31;
    const int col  = lane & 15;
    const int rh   = (lane >> 4) << 3;
#pragma unroll
    for (int i = 0; i < 4; ++i)
#pragma unroll
        for (int j = 0; j < 4; ++j)
#pragma unroll
            for (int r = 0; r < 8; ++r) {
                const int m = M0 + i * 16 + rh + r;
                const int n = N0 + j * 16 + col;
                const float val = acc[i][j][r] + bias[n];
                const int bb = m >> 11, s = m & (SEQ - 1);
                const int h  = n >> 6,  d = n & (D_K - 1);
                if (MODE == 0)
                    out[(((size_t)(bb * N_HEAD + h)) * SEQ + s) * D_K + d] = (__bf16)val;
                else
                    out[(((size_t)(bb * N_HEAD + h)) * D_K + d) * SEQ + s] = (__bf16)val;
            }
}

// ---------------------------------------------------------------------------
// Scores: attn_raw[bh, m, n] = (1/8) * sum_d Q[bh,m,d] * K[bh,n,d]
// Per (b,h): M=N=2048, K=64 (only 2 k-steps; no LDS staging needed).
// ---------------------------------------------------------------------------
__global__ void __launch_bounds__(256)
scores_kernel(const __bf16* __restrict__ Qh, const __bf16* __restrict__ Kh,
              float* __restrict__ attn) {
    const int bh   = blockIdx.y;
    const int wave = threadIdx.x >> 5;
    const int t    = blockIdx.x * 8 + wave;       // 32*32 = 1024 tiles per bh
    const int tm   = t >> 5, tn = t & 31;
    const int M0 = tm * 64, N0 = tn * 64;

    const __bf16* Qb = Qh + (size_t)bh * SEQ * D_K;
    const __bf16* Kb = Kh + (size_t)bh * SEQ * D_K;

    v8f acc[4][4] = {};
#pragma unroll
    for (int k = 0; k < D_K; k += 32) {
        v16bf a[4], b[4];
#pragma unroll
        for (int i = 0; i < 4; ++i)
            a[i] = load_frag_bf16(Qb + (size_t)(M0 + i * 16) * D_K + k, D_K);
#pragma unroll
        for (int j = 0; j < 4; ++j)
            b[j] = load_frag_bf16(Kb + (size_t)(N0 + j * 16) * D_K + k, D_K);
#pragma unroll
        for (int i = 0; i < 4; ++i)
#pragma unroll
            for (int j = 0; j < 4; ++j)
                acc[i][j] = wmma_bf16(a[i], b[j], acc[i][j]);
    }

    float* Ab = attn + (size_t)bh * SEQ * SEQ;
    const int lane = threadIdx.x & 31;
    const int col  = lane & 15;
    const int rh   = (lane >> 4) << 3;
#pragma unroll
    for (int i = 0; i < 4; ++i)
#pragma unroll
        for (int j = 0; j < 4; ++j)
#pragma unroll
            for (int r = 0; r < 8; ++r) {
                const int m = M0 + i * 16 + rh + r;
                const int n = N0 + j * 16 + col;
                Ab[(size_t)m * SEQ + n] = acc[i][j][r] * 0.125f;  // 1/sqrt(64)
            }
}

// ---------------------------------------------------------------------------
// Row softmax in place over the last dim (2048).  One 256-thread block / row.
// ---------------------------------------------------------------------------
__global__ void __launch_bounds__(256)
softmax_kernel(float* __restrict__ attn) {
    __shared__ float red[256];
    float* p = attn + (size_t)blockIdx.x * SEQ;
    const int t = threadIdx.x;

    float vals[8];
    float m = -3.4e38f;
#pragma unroll
    for (int i = 0; i < 8; ++i) { vals[i] = p[t + i * 256]; m = fmaxf(m, vals[i]); }
    red[t] = m;
    __syncthreads();
    for (int s = 128; s > 0; s >>= 1) {
        if (t < s) red[t] = fmaxf(red[t], red[t + s]);
        __syncthreads();
    }
    m = red[0];
    __syncthreads();

    float sum = 0.f;
#pragma unroll
    for (int i = 0; i < 8; ++i) { vals[i] = __expf(vals[i] - m); sum += vals[i]; }
    red[t] = sum;
    __syncthreads();
    for (int s = 128; s > 0; s >>= 1) {
        if (t < s) red[t] += red[t + s];
        __syncthreads();
    }
    const float inv = 1.0f / red[0];
#pragma unroll
    for (int i = 0; i < 8; ++i) p[t + i * 256] = vals[i] * inv;
}

// ---------------------------------------------------------------------------
// Context: ctx[bh, s, d] = sum_j P[bh, s, j] * Vt[bh, d, j]
// Per (b,h): M=2048, N=64, K=2048.  All 8 waves of a block share the same
// Vt slab -> async double-buffered LDS staging over the 64-step K loop.
// ---------------------------------------------------------------------------
__global__ void __launch_bounds__(256)
ctx_kernel(const float* __restrict__ attn, const __bf16* __restrict__ Vt,
           __bf16* __restrict__ ctxb) {
    __shared__ __bf16 smemB[2][64 * 32];

    const int bh   = blockIdx.y;
    const int wave = threadIdx.x >> 5;
    const int M0   = blockIdx.x * 512 + wave * 64;

    const float*  Pb = attn + (size_t)bh * SEQ * SEQ;
    const __bf16* Vb = Vt   + (size_t)bh * D_K * SEQ;
    const int NK = SEQ / 32;                       // 64 k-steps

    stage_slab(Vb, SEQ, 0, smemB[0]);
    wait_async0();
    __syncthreads();

    v8f acc[4][4] = {};
    for (int kk = 0; kk < NK; ++kk) {
        const int cur = kk & 1;
        if (kk + 1 < NK) stage_slab(Vb, SEQ, (kk + 1) * 32, smemB[1 - cur]);

        const int k = kk * 32;
        v16bf a[4], b[4];
#pragma unroll
        for (int i = 0; i < 4; ++i)
            a[i] = load_frag_f32(Pb + (size_t)(M0 + i * 16) * SEQ + k, SEQ);
#pragma unroll
        for (int j = 0; j < 4; ++j)
            b[j] = load_frag_smem(&smemB[cur][j * 16 * 32]);
#pragma unroll
        for (int i = 0; i < 4; ++i)
#pragma unroll
            for (int j = 0; j < 4; ++j)
                acc[i][j] = wmma_bf16(a[i], b[j], acc[i][j]);

        if (kk + 1 < NK) { wait_async0(); __syncthreads(); }
    }

    const int bb = bh >> 4, h = bh & 15;
    const int lane = threadIdx.x & 31;
    const int col  = lane & 15;
    const int rh   = (lane >> 4) << 3;
#pragma unroll
    for (int i = 0; i < 4; ++i)
#pragma unroll
        for (int j = 0; j < 4; ++j)
#pragma unroll
            for (int r = 0; r < 8; ++r) {
                const int s = M0 + i * 16 + rh + r;
                const int d = j * 16 + col;
                ctxb[((size_t)(bb * SEQ + s)) * D_MODEL + h * D_K + d] = (__bf16)acc[i][j][r];
            }
}

// ---------------------------------------------------------------------------
// Output projection: out[m,n] = sum_k ctx[m,k] * Wo[n,k] + bo[n]  (f32 out)
// Same block structure / async weight staging as proj_kernel.
// ---------------------------------------------------------------------------
__global__ void __launch_bounds__(256)
outproj_kernel(const __bf16* __restrict__ ctxb, const __bf16* __restrict__ Wo,
               const float* __restrict__ bo, float* __restrict__ out) {
    __shared__ __bf16 smemB[2][64 * 32];

    const int wave   = threadIdx.x >> 5;
    const int blockM = blockIdx.x >> 4;
    const int blockN = blockIdx.x & 15;
    const int M0 = blockM * 512 + wave * 64;
    const int N0 = blockN * 64;

    const __bf16* Wb = Wo + (size_t)N0 * D_MODEL;
    const int NK = D_MODEL / 32;

    stage_slab(Wb, D_MODEL, 0, smemB[0]);
    wait_async0();
    __syncthreads();

    v8f acc[4][4] = {};
    for (int kk = 0; kk < NK; ++kk) {
        const int cur = kk & 1;
        if (kk + 1 < NK) stage_slab(Wb, D_MODEL, (kk + 1) * 32, smemB[1 - cur]);

        const int k = kk * 32;
        v16bf a[4], b[4];
#pragma unroll
        for (int i = 0; i < 4; ++i)
            a[i] = load_frag_bf16(ctxb + (size_t)(M0 + i * 16) * D_MODEL + k, D_MODEL);
#pragma unroll
        for (int j = 0; j < 4; ++j)
            b[j] = load_frag_smem(&smemB[cur][j * 16 * 32]);
#pragma unroll
        for (int i = 0; i < 4; ++i)
#pragma unroll
            for (int j = 0; j < 4; ++j)
                acc[i][j] = wmma_bf16(a[i], b[j], acc[i][j]);

        if (kk + 1 < NK) { wait_async0(); __syncthreads(); }
    }

    const int lane = threadIdx.x & 31;
    const int col  = lane & 15;
    const int rh   = (lane >> 4) << 3;
#pragma unroll
    for (int i = 0; i < 4; ++i)
#pragma unroll
        for (int j = 0; j < 4; ++j)
#pragma unroll
            for (int r = 0; r < 8; ++r) {
                const int m = M0 + i * 16 + rh + r;
                const int n = N0 + j * 16 + col;
                out[(size_t)m * D_MODEL + n] = acc[i][j][r] + bo[n];
            }
}

// ---------------------------------------------------------------------------
// Host launcher
// ---------------------------------------------------------------------------
extern "C" void kernel_launch(void* const* d_in, const int* in_sizes, int n_in,
                              void* d_out, int out_size, void* d_ws, size_t ws_size,
                              hipStream_t stream) {
    const float* q  = (const float*)d_in[0];
    const float* k  = (const float*)d_in[1];
    const float* v  = (const float*)d_in[2];
    const float* wq = (const float*)d_in[3];
    const float* bq = (const float*)d_in[4];
    const float* wk = (const float*)d_in[5];
    const float* bk = (const float*)d_in[6];
    const float* wv = (const float*)d_in[7];
    const float* bv = (const float*)d_in[8];
    const float* wo = (const float*)d_in[9];
    const float* bo = (const float*)d_in[10];

    // Workspace carve-up (bf16 staging; 64 MB total).
    char* w = (char*)d_ws;
    const size_t ACT = (size_t)ROWS * D_MODEL;        // 4096*1024
    const size_t WSZ = (size_t)D_MODEL * D_MODEL;     // 1024*1024
    __bf16* qb   = (__bf16*)w; w += ACT * 2;
    __bf16* kb   = (__bf16*)w; w += ACT * 2;
    __bf16* vb   = (__bf16*)w; w += ACT * 2;
    __bf16* wqb  = (__bf16*)w; w += WSZ * 2;
    __bf16* wkb  = (__bf16*)w; w += WSZ * 2;
    __bf16* wvb  = (__bf16*)w; w += WSZ * 2;
    __bf16* wob  = (__bf16*)w; w += WSZ * 2;
    __bf16* Qh   = (__bf16*)w; w += ACT * 2;          // [B,H,S,dk]
    __bf16* Kh   = (__bf16*)w; w += ACT * 2;          // [B,H,S,dk]
    __bf16* Vt   = (__bf16*)w; w += ACT * 2;          // [B,H,dk,S]
    __bf16* ctxb = (__bf16*)w; w += ACT * 2;          // [B,S,D]

    float* out  = (float*)d_out;
    float* attn = out + ACT;                          // attn region of d_out

    // 1) bf16 staging
    convert_kernel<<<1024, 256, 0, stream>>>(q,  qb,  (int)ACT);
    convert_kernel<<<1024, 256, 0, stream>>>(k,  kb,  (int)ACT);
    convert_kernel<<<1024, 256, 0, stream>>>(v,  vb,  (int)ACT);
    convert_kernel<<<256,  256, 0, stream>>>(wq, wqb, (int)WSZ);
    convert_kernel<<<256,  256, 0, stream>>>(wk, wkb, (int)WSZ);
    convert_kernel<<<256,  256, 0, stream>>>(wv, wvb, (int)WSZ);
    convert_kernel<<<256,  256, 0, stream>>>(wo, wob, (int)WSZ);

    // 2) projections (8 M-blocks x 16 N-blocks = 128 blocks)
    proj_kernel<0><<<128, 256, 0, stream>>>(qb, wqb, bq, Qh);
    proj_kernel<0><<<128, 256, 0, stream>>>(kb, wkb, bk, Kh);
    proj_kernel<1><<<128, 256, 0, stream>>>(vb, wvb, bv, Vt);

    // 3) scores -> attn region (32 bh pairs x 1024 wave-tiles)
    scores_kernel<<<dim3(128, 32), 256, 0, stream>>>(Qh, Kh, attn);

    // 4) softmax in place (B*H*S = 65536 rows)
    softmax_kernel<<<65536, 256, 0, stream>>>(attn);

    // 5) context (4 M-blocks x 32 bh pairs)
    ctx_kernel<<<dim3(4, 32), 256, 0, stream>>>(attn, Vt, ctxb);

    // 6) output projection -> out region
    outproj_kernel<<<128, 256, 0, stream>>>(ctxb, wob, bo, out);
}
